// SparseConvFCN_89232240542136
// MI455X (gfx1250) — matrix operand order repaired and checked by reading, not compiled, not run on hardware
//
#include <hip/hip_runtime.h>
#include <hip/hip_bf16.h>

// ---------------------------------------------------------------------------
// CDNA5 (gfx1250) sparse-conv U-Net, f16 WMMA (16x16x32, f32 accum).
// Each wave computes a 16-site x 64-col output tile (4 WMMA accumulators), so
// one gathered A fragment feeds 4 v_wmma ops.
//
// Assumed d_in order (dict order; 'params' pytree flattened with sorted keys):
//   0: x [N0,32]
//   1: W_init [27,32,64]
//   2+3l: down[l].W [8,Ca,Cb]   3+3l: down[l].b   4+3l: down[l].g      (l=0..2)
//   11+6l: lev[l].W1 [27,C,C]  12+6l: W2  13+6l: b1  14+6l: b2  15+6l: g1  16+6l: g2
//   35: lin_W [640,32]   36: lin_b [32]
//   37..40: nbr0..nbr3 [N_l,27]  (pad index == N_l)
//   41+2l: parent_l [N_l]   42+2l: off_l [N_l]
// Output: [N0,32] fp32.
// ---------------------------------------------------------------------------

typedef _Float16 f16_t;
typedef __attribute__((ext_vector_type(16))) _Float16 v16h;
typedef __attribute__((ext_vector_type(8)))  _Float16 v8h;
typedef __attribute__((ext_vector_type(8)))  float    v8f;

#define WPB 4  // waves per block in WMMA kernels (128 threads)
#define NTW 4  // cout tiles per wave (64 output columns)

// ---- pack fp32 weights [K,Cin,Cout] into WMMA B-fragment layout -----------
// frag f = (k*cinChunks + ck)*coutTiles + nt ; element lane*16+e holds
// W[k, ck*32 + (lane/16)*16 + e, nt*16 + lane%16]   (B: K=(lane/16)*16+e, N=lane%16)
__global__ void k_pack_w(const float* __restrict__ W, f16_t* __restrict__ dst,
                         int K, int Cin, int Cout) {
  size_t total = (size_t)K * Cin * Cout;
  int cinChunks = Cin >> 5, coutTiles = Cout >> 4;
  for (size_t i = (size_t)blockIdx.x * blockDim.x + threadIdx.x; i < total;
       i += (size_t)gridDim.x * blockDim.x) {
    size_t frag = i >> 9;
    int le = (int)(i & 511);
    int lane = le >> 4, e = le & 15;
    int nt = (int)(frag % coutTiles);
    size_t t = frag / coutTiles;
    int ck = (int)(t % cinChunks);
    int k  = (int)(t / cinChunks);
    int kpos = ck * 32 + ((lane >> 4) << 4) + e;
    int ncol = nt * 16 + (lane & 15);
    dst[i] = (f16_t)W[((size_t)k * Cin + kpos) * Cout + ncol];
  }
}

// ---- BN statistics: per-channel sum & sumsq via atomics (stats pre-zeroed) -
__global__ void k_bn_reduce(const float* __restrict__ x, float* __restrict__ stats,
                            int N, int C) {
  int c = threadIdx.x;  // blockDim.x == C (<=256)
  float s = 0.f, s2 = 0.f;
  for (int n = blockIdx.x; n < N; n += gridDim.x) {
    float v = x[(size_t)n * C + c];
    s += v; s2 += v * v;
  }
  atomicAdd(&stats[c], s);
  atomicAdd(&stats[C + c], s2);
}

// ---- fused BN + ReLU + f32->f16, writes zero pad row at index N -----------
__global__ void k_bn_apply_f16(const float* __restrict__ x, const float* __restrict__ stats,
                               const float* __restrict__ g, const float* __restrict__ b,
                               f16_t* __restrict__ y, int N, int C) {
  size_t total = (size_t)(N + 1) * C;
  float invN = 1.f / (float)N;
  for (size_t i = (size_t)blockIdx.x * blockDim.x + threadIdx.x; i < total;
       i += (size_t)gridDim.x * blockDim.x) {
    size_t n = i / C;
    int c = (int)(i - n * C);
    if (n >= (size_t)N) { y[i] = (f16_t)0.0f; continue; }
    float mu  = stats[c] * invN;
    float var = stats[C + c] * invN - mu * mu;
    float k   = g[c] * rsqrtf(var + 1e-4f);
    float v   = (x[i] - mu) * k + b[c];
    y[i] = (f16_t)fmaxf(v, 0.f);
  }
}

// ---- plain f32->f16 (with optional pad row) --------------------------------
__global__ void k_cvt_f16(const float* __restrict__ x, f16_t* __restrict__ y,
                          int N, int C, int pad) {
  size_t total = (size_t)(N + pad) * C;
  size_t nc = (size_t)N * C;
  for (size_t i = (size_t)blockIdx.x * blockDim.x + threadIdx.x; i < total;
       i += (size_t)gridDim.x * blockDim.x)
    y[i] = (i < nc) ? (f16_t)x[i] : (f16_t)0.0f;
}

// ---- parent-chain composition ---------------------------------------------
__global__ void k_compose(const int* __restrict__ p, const int* __restrict__ idx,
                          int* __restrict__ out, int N) {
  for (int n = blockIdx.x * blockDim.x + threadIdx.x; n < N;
       n += gridDim.x * blockDim.x)
    out[n] = p[idx[n]];
}

// ---- A fragment loader: 16-bit A 16x32 layout ------------------------------
// lane<16: e0..15 -> K=0..15 ; lane>=16: K=16..31
__device__ __forceinline__ v16h load_a_frag(const f16_t* row, int lane) {
  int koff = (lane >> 4) << 4;
  v8h lo = *(const v8h*)(row + koff);
  v8h hi = *(const v8h*)(row + 8 + koff);
  v16h a;
#pragma unroll
  for (int i = 0; i < 8; ++i) { a[i] = lo[i]; a[8 + i] = hi[i]; }
  return a;
}

// ---- submanifold 3^3 conv: gather f16 rows, WMMA over 27*(Cin/32) ----------
// One wave: 16 sites x (NTW*16) output columns; A fragment reused NTW times.
__global__ __launch_bounds__(32 * WPB)
void k_smconv(const f16_t* __restrict__ x16,   // (N+1) x Cin, row N == zeros
              const int* __restrict__ nbr,     // N x 27, pad == N
              const f16_t* __restrict__ wpack, // packed B fragments
              float* __restrict__ out,         // N x Cout
              int N, int Cin, int Cout, int accumulate) {
  int lane = threadIdx.x & 31;
  int wave = threadIdx.x >> 5;
  int mt = blockIdx.x * WPB + wave;           // site tile
  int ntBase = blockIdx.y * NTW;              // first cout tile
  int coutTiles = Cout >> 4;
  int cinChunks = Cin >> 5;
  int row = mt * 16 + (lane & 15);
  bool rowValid = row < N;

  v8f acc[NTW];
#pragma unroll
  for (int t = 0; t < NTW; ++t) acc[t] = (v8f){};
  if (accumulate) {
#pragma unroll
    for (int t = 0; t < NTW; ++t) {
      int col = (ntBase + t) * 16 + (lane & 15);
#pragma unroll
      for (int v = 0; v < 8; ++v) {
        int srow = mt * 16 + v + ((lane >> 4) << 3);
        acc[t][v] = (srow < N) ? out[(size_t)srow * Cout + col] : 0.f;
      }
    }
  }

  for (int k = 0; k < 27; ++k) {
    int r = rowValid ? nbr[(size_t)row * 27 + k] : N;   // pad row -> zeros
    const f16_t* arow = x16 + (size_t)r * Cin;
    const f16_t* wk = wpack + ((size_t)k * cinChunks) * coutTiles * 512;
    for (int ck = 0; ck < cinChunks; ++ck) {
      v16h a = load_a_frag(arow + ck * 32, lane);
      const f16_t* wck = wk + ((size_t)ck * coutTiles + ntBase) * 512 + lane * 16;
#pragma unroll
      for (int t = 0; t < NTW; ++t) {
        v16h b = *(const v16h*)(wck + (size_t)t * 512);
        acc[t] = __builtin_amdgcn_wmma_f32_16x16x32_f16(false, a, false, b,
                                                        (short)0, acc[t], false, false);
      }
    }
  }

#pragma unroll
  for (int t = 0; t < NTW; ++t) {
    int col = (ntBase + t) * 16 + (lane & 15);
#pragma unroll
    for (int v = 0; v < 8; ++v) {
      int srow = mt * 16 + v + ((lane >> 4) << 3);
      if (srow < N) out[(size_t)srow * Cout + col] = acc[t][v];
    }
  }
}

// ---- stride-2 filter-2 down-conv: per-row offset mask, WMMA, scatter-add ---
// A chunk loaded once per ck, masked 8x (VALU selects) against the 8 weights.
__global__ __launch_bounds__(32 * WPB)
void k_downconv(const f16_t* __restrict__ x16,   // (N+1) x Cin (BN+ReLU'd)
                const int* __restrict__ off, const int* __restrict__ parent,
                const f16_t* __restrict__ wpack, // [8,Cin,Cout] packed
                float* __restrict__ out,         // Nout x Cout, pre-zeroed
                int N, int Cin, int Cout) {
  int lane = threadIdx.x & 31;
  int wave = threadIdx.x >> 5;
  int mt = blockIdx.x * WPB + wave;
  int ntBase = blockIdx.y * NTW;
  int coutTiles = Cout >> 4;
  int cinChunks = Cin >> 5;
  int row = mt * 16 + (lane & 15);
  bool rowValid = row < N;
  int rowOff = rowValid ? off[row] : -1;
  const f16_t* arow = x16 + (size_t)(rowValid ? row : 0) * Cin;

  v8f acc[NTW];
#pragma unroll
  for (int t = 0; t < NTW; ++t) acc[t] = (v8f){};

  for (int ck = 0; ck < cinChunks; ++ck) {
    v16h a = load_a_frag(arow + ck * 32, lane);
    for (int o = 0; o < 8; ++o) {
      bool match = (rowOff == o);
      v16h am;
#pragma unroll
      for (int i = 0; i < 16; ++i) am[i] = match ? a[i] : (f16_t)0.0f;
      const f16_t* wo = wpack +
          ((size_t)(o * cinChunks + ck) * coutTiles + ntBase) * 512 + lane * 16;
#pragma unroll
      for (int t = 0; t < NTW; ++t) {
        v16h b = *(const v16h*)(wo + (size_t)t * 512);
        acc[t] = __builtin_amdgcn_wmma_f32_16x16x32_f16(false, am, false, b,
                                                        (short)0, acc[t], false, false);
      }
    }
  }

#pragma unroll
  for (int t = 0; t < NTW; ++t) {
    int col = (ntBase + t) * 16 + (lane & 15);
#pragma unroll
    for (int v = 0; v < 8; ++v) {
      int srow = mt * 16 + v + ((lane >> 4) << 3);
      if (srow < N)
        atomicAdd(&out[(size_t)parent[srow] * Cout + col], acc[t][v]);
    }
  }
}

// ---- final linear over implicit concat [h0 | h1[q1] | h2[q2] | h3[q3]] -----
__global__ __launch_bounds__(32 * WPB)
void k_final_linear(const f16_t* __restrict__ h0, const f16_t* __restrict__ h1,
                    const f16_t* __restrict__ h2, const f16_t* __restrict__ h3,
                    const int* __restrict__ q1, const int* __restrict__ q2,
                    const int* __restrict__ q3,
                    const f16_t* __restrict__ wpack,  // 20 chunks x 2 tiles x 512
                    const float* __restrict__ bias,
                    float* __restrict__ out, int N0) {
  int lane = threadIdx.x & 31;
  int wave = threadIdx.x >> 5;
  int mt = blockIdx.x * WPB + wave;
  int row = mt * 16 + (lane & 15);
  bool rowValid = row < N0;

  v8f acc[2];
  acc[0] = (v8f){}; acc[1] = (v8f){};
  for (int j = 0; j < 20; ++j) {      // K chunks of 32 over 640 = 64+128+192+256
    const f16_t* hl; int C, base, r;
    if (j < 2)       { hl = h0; C = 64;  base = j * 32;        r = rowValid ? row     : 0; }
    else if (j < 6)  { hl = h1; C = 128; base = (j - 2) * 32;  r = rowValid ? q1[row] : 0; }
    else if (j < 12) { hl = h2; C = 192; base = (j - 6) * 32;  r = rowValid ? q2[row] : 0; }
    else             { hl = h3; C = 256; base = (j - 12) * 32; r = rowValid ? q3[row] : 0; }
    v16h a = load_a_frag(hl + (size_t)r * C + base, lane);
#pragma unroll
    for (int t = 0; t < 2; ++t) {
      v16h b = *(const v16h*)(wpack + ((size_t)(j * 2 + t)) * 512 + lane * 16);
      acc[t] = __builtin_amdgcn_wmma_f32_16x16x32_f16(false, a, false, b,
                                                      (short)0, acc[t], false, false);
    }
  }

#pragma unroll
  for (int t = 0; t < 2; ++t) {
    int col = t * 16 + (lane & 15);
    float bs = bias[col];
#pragma unroll
    for (int v = 0; v < 8; ++v) {
      int srow = mt * 16 + v + ((lane >> 4) << 3);
      if (srow < N0) out[(size_t)srow * 32 + col] = acc[t][v] + bs;
    }
  }
}

// ---------------------------------------------------------------------------
extern "C" void kernel_launch(void* const* d_in, const int* in_sizes, int n_in,
                              void* d_out, int out_size, void* d_ws, size_t ws_size,
                              hipStream_t stream) {
  (void)n_in; (void)out_size;
  const int C[4] = {64, 128, 192, 256};
  int N[4];
  for (int l = 0; l < 4; ++l) N[l] = in_sizes[37 + l] / 27;
  const int N0 = N[0];

  // input pointers
  const float* x      = (const float*)d_in[0];
  const float* W_init = (const float*)d_in[1];
  const float* dW[3], *db[3], *dg[3];
  for (int l = 0; l < 3; ++l) {
    dW[l] = (const float*)d_in[2 + 3 * l];
    db[l] = (const float*)d_in[3 + 3 * l];
    dg[l] = (const float*)d_in[4 + 3 * l];
  }
  const float *W1[4], *W2[4], *b1[4], *b2[4], *g1[4], *g2[4];
  for (int l = 0; l < 4; ++l) {
    W1[l] = (const float*)d_in[11 + 6 * l];
    W2[l] = (const float*)d_in[12 + 6 * l];
    b1[l] = (const float*)d_in[13 + 6 * l];
    b2[l] = (const float*)d_in[14 + 6 * l];
    g1[l] = (const float*)d_in[15 + 6 * l];
    g2[l] = (const float*)d_in[16 + 6 * l];
  }
  const float* lin_W = (const float*)d_in[35];
  const float* lin_b = (const float*)d_in[36];
  const int* nbr[4];   for (int l = 0; l < 4; ++l) nbr[l]    = (const int*)d_in[37 + l];
  const int* parent[3]; const int* off[3];
  for (int l = 0; l < 3; ++l) { parent[l] = (const int*)d_in[41 + 2 * l];
                                off[l]    = (const int*)d_in[42 + 2 * l]; }
  float* outp = (float*)d_out;

  // ---- carve workspace (256B aligned chunks) ----
  size_t wsoff = 0;
  auto carve = [&](size_t bytes) -> void* {
    void* p = (char*)d_ws + wsoff;
    wsoff += (bytes + 255) & ~(size_t)255;
    return p;
  };
  f16_t* wpInit = (f16_t*)carve((size_t)27 * 32 * 64 * 2);
  f16_t *wpW1[4], *wpW2[4], *wpD[3];
  for (int l = 0; l < 4; ++l) {
    wpW1[l] = (f16_t*)carve((size_t)27 * C[l] * C[l] * 2);
    wpW2[l] = (f16_t*)carve((size_t)27 * C[l] * C[l] * 2);
  }
  for (int l = 0; l < 3; ++l) wpD[l] = (f16_t*)carve((size_t)8 * C[l] * C[l + 1] * 2);
  f16_t* wpLin = (f16_t*)carve((size_t)640 * 32 * 2);
  float* h[4]; f16_t* h16[4];
  for (int l = 0; l < 4; ++l) h[l]   = (float*)carve((size_t)N[l] * C[l] * 4);
  for (int l = 0; l < 4; ++l) h16[l] = (f16_t*)carve((size_t)N[l] * C[l] * 2);
  f16_t* x16 = (f16_t*)carve((size_t)(N0 + 1) * 32 * 2);
  size_t maxXhat = 0, maxY = 0;
  for (int l = 0; l < 4; ++l) {
    size_t a = (size_t)(N[l] + 1) * C[l], b = (size_t)N[l] * C[l];
    if (a > maxXhat) maxXhat = a;
    if (b > maxY) maxY = b;
  }
  f16_t* xhat  = (f16_t*)carve(maxXhat * 2);
  float* ybuf  = (float*)carve(maxY * 4);
  float* stats = (float*)carve(512 * 4);
  int* q2 = (int*)carve((size_t)N0 * 4);
  int* q3 = (int*)carve((size_t)N0 * 4);
  (void)ws_size;

  auto gsz = [](size_t total, int blk) -> int {
    size_t g = (total + blk - 1) / blk;
    return (int)(g > 4096 ? 4096 : g);
  };

  // ---- pack all weights to f16 B-fragment layout ----
  k_pack_w<<<gsz((size_t)27 * 32 * 64, 256), 256, 0, stream>>>(W_init, wpInit, 27, 32, 64);
  for (int l = 0; l < 4; ++l) {
    size_t t = (size_t)27 * C[l] * C[l];
    k_pack_w<<<gsz(t, 256), 256, 0, stream>>>(W1[l], wpW1[l], 27, C[l], C[l]);
    k_pack_w<<<gsz(t, 256), 256, 0, stream>>>(W2[l], wpW2[l], 27, C[l], C[l]);
  }
  for (int l = 0; l < 3; ++l)
    k_pack_w<<<gsz((size_t)8 * C[l] * C[l + 1], 256), 256, 0, stream>>>(
        dW[l], wpD[l], 8, C[l], C[l + 1]);
  k_pack_w<<<gsz((size_t)640 * 32, 256), 256, 0, stream>>>(lin_W, wpLin, 1, 640, 32);

  // ---- initial conv: x (f16, padded) -> h0 ----
  k_cvt_f16<<<gsz((size_t)(N0 + 1) * 32, 256), 256, 0, stream>>>(x, x16, N0, 32, 1);
  {
    int siteTiles = (N0 + 15) / 16;
    dim3 grid((siteTiles + WPB - 1) / WPB, (64 / 16) / NTW);
    k_smconv<<<grid, 32 * WPB, 0, stream>>>(x16, nbr[0], wpInit, h[0], N0, 32, 64, 0);
  }

  // ---- per-level residual blocks + down path ----
  for (int l = 0; l < 4; ++l) {
    int Nl = N[l], Cl = C[l];
    int siteTiles = (Nl + 15) / 16;
    dim3 grid((siteTiles + WPB - 1) / WPB, (Cl / 16) / NTW);

    hipMemsetAsync(stats, 0, 512 * 4, stream);
    k_bn_reduce<<<256, Cl, 0, stream>>>(h[l], stats, Nl, Cl);
    k_bn_apply_f16<<<gsz((size_t)(Nl + 1) * Cl, 256), 256, 0, stream>>>(
        h[l], stats, g1[l], b1[l], xhat, Nl, Cl);
    k_smconv<<<grid, 32 * WPB, 0, stream>>>(xhat, nbr[l], wpW1[l], ybuf, Nl, Cl, Cl, 0);

    hipMemsetAsync(stats, 0, 512 * 4, stream);
    k_bn_reduce<<<256, Cl, 0, stream>>>(ybuf, stats, Nl, Cl);
    k_bn_apply_f16<<<gsz((size_t)(Nl + 1) * Cl, 256), 256, 0, stream>>>(
        ybuf, stats, g2[l], b2[l], xhat, Nl, Cl);
    k_smconv<<<grid, 32 * WPB, 0, stream>>>(xhat, nbr[l], wpW2[l], h[l], Nl, Cl, Cl, 1);

    if (l < 3) {
      int Cn = C[l + 1], Nn = N[l + 1];
      hipMemsetAsync(stats, 0, 512 * 4, stream);
      k_bn_reduce<<<256, Cl, 0, stream>>>(h[l], stats, Nl, Cl);
      k_bn_apply_f16<<<gsz((size_t)(Nl + 1) * Cl, 256), 256, 0, stream>>>(
          h[l], stats, dg[l], db[l], xhat, Nl, Cl);
      hipMemsetAsync(h[l + 1], 0, (size_t)Nn * Cn * 4, stream);
      dim3 gridd((siteTiles + WPB - 1) / WPB, (Cn / 16) / NTW);
      k_downconv<<<gridd, 32 * WPB, 0, stream>>>(xhat, off[l], parent[l], wpD[l],
                                                 h[l + 1], Nl, Cl, Cn);
    }
  }

  // ---- final linear over implicit concat ----
  for (int l = 0; l < 4; ++l)
    k_cvt_f16<<<gsz((size_t)N[l] * C[l], 256), 256, 0, stream>>>(h[l], h16[l], N[l], C[l], 0);
  k_compose<<<gsz(N0, 256), 256, 0, stream>>>(parent[1], parent[0], q2, N0);
  k_compose<<<gsz(N0, 256), 256, 0, stream>>>(parent[2], q2, q3, N0);
  {
    int siteTiles = (N0 + 15) / 16;
    dim3 grid((siteTiles + WPB - 1) / WPB, 1);
    k_final_linear<<<grid, 32 * WPB, 0, stream>>>(h16[0], h16[1], h16[2], h16[3],
                                                  parent[0], q2, q3, wpLin, lin_b,
                                                  outp, N0);
  }
}
